// BiGatedDecoderCA_91001767068338
// MI455X (gfx1250) — matrix operand order repaired and checked by reading, not compiled, not run on hardware
//
#include <hip/hip_runtime.h>
#include <hip/hip_bf16.h>

// ---------------------------------------------------------------------------
// MI455X / gfx1250 implementation of BiGatedDecoderCA.
// All heavy math runs through v_wmma_f32_16x16x32_bf16 (bf16 in, f32 accum).
// BN folded into LDS staging; residual/gamma/delta/GELU folded into GEMM
// epilogues.  f32 softmax.  Memory-bound op => minimize passes over HBM.
// Round 4: sched_barrier pins all fragment ds_loads before the WMMA burst
// (partial dscnt waits, disjoint frag registers); HAS_SCALE=false staging no
// longer emits x*1+0 (v_pk_add_f32 ...,0) dead VALU.
// ---------------------------------------------------------------------------

typedef __bf16 bf16_t;
typedef __attribute__((ext_vector_type(16))) __bf16 v16bf;
typedef __attribute__((ext_vector_type(8)))  __bf16 v8bf;
typedef __attribute__((ext_vector_type(8)))  float  v8f;

#define HW_N  12544     // 112*112, == 98 * 128 exactly
#define IMG_W 112
#define NB    8

#define SCHED_FENCE() __builtin_amdgcn_sched_barrier(0)

__device__ __forceinline__ v16bf join8(v8bf lo, v8bf hi) {
  union { v16bf v; v8bf h[2]; } u;
  u.h[0] = lo; u.h[1] = hi;
  return u.v;
}

__device__ __forceinline__ unsigned pack_bf16x2(float lo, float hi) {
  union { bf16_t h[2]; unsigned u; } pk;
  pk.h[0] = (bf16_t)lo; pk.h[1] = (bf16_t)hi;
  return pk.u;
}

// ---------------------------------------------------------------------------
// Generic 1x1-conv GEMM:  Out[b][m][n] = sum_k W[m][k] * (X[b][k][n]*sc[k]+bi[k])
// Block = 256 thr (8 waves), tile 64(M) x 128(N), K pipelined 32 at a time
// through a double-buffered LDS stage with register prefetch.
// Wave w: rows [16*(w/2), +16), cols [64*(w&1), +64)  -> 4 WMMA tiles / k-step.
// A fragment: lane(l&15)=row, l>>4 selects K-half; two contiguous 16B LDS loads.
// B staged transposed (Bt[n][k]); each thread stages a k-pair for 8 columns so
// every LDS write is one packed b32.
// MODE: 0 = plain store, 1 = res + alpha*acc, 2 = res + alpha*gelu(acc)
// ---------------------------------------------------------------------------
template<int MODE, bool HAS_SCALE>
__global__ void __launch_bounds__(256)
gemm_conv1_wmma(const float* __restrict__ W, const float* __restrict__ X,
                float* __restrict__ Out,
                const float* __restrict__ xScale, const float* __restrict__ xBias,
                const float* __restrict__ res, const float* __restrict__ alphaPtr,
                int Kdim, long xBatchStride, long outBatchStride,
                long resBatchStride)
{
  const int N = HW_N;
  struct Tiles { bf16_t A[64][32]; bf16_t B[128][32]; };
  __shared__ __align__(16) Tiles tiles[2];

  const int b    = blockIdx.z;
  const int m0   = blockIdx.y * 64;
  const int n0   = blockIdx.x * 128;
  const int tid  = threadIdx.x;
  const int wave = tid >> 5;
  const int lane = tid & 31;
  const int wm   = wave >> 1;     // 0..3  (M tile)
  const int wn   = wave & 1;      // 0..1  (64-col half)
  const int rl   = lane & 15;
  const int kh   = lane >> 4;

  // staging coordinates (loop-invariant)
  const int am  = tid >> 2;           // A: row 0..63
  const int akg = (tid & 3) << 3;     // A: k group 0,8,16,24
  const int kp  = tid >> 4;           // B: k-pair 0..15  (k = 2kp, 2kp+1)
  const int ng  = (tid & 15) << 3;    // B: col group 0..120

  const float* baseA = W + (size_t)(m0 + am) * Kdim + akg;
  const float* baseB = X + (size_t)b * xBatchStride + (size_t)(2 * kp) * N + n0 + ng;

  // write prefetched registers (for absolute k-offset kOff) into LDS buffer
  auto store_tile = [&](int buf, int kOff, const float* pa,
                        const float* pb0, const float* pb1) {
    #pragma unroll
    for (int i = 0; i < 8; ++i) tiles[buf].A[am][akg + i] = (bf16_t)pa[i];
    if (HAS_SCALE) {
      const float sc0 = xScale[kOff + 2 * kp];     const float bi0 = xBias[kOff + 2 * kp];
      const float sc1 = xScale[kOff + 2 * kp + 1]; const float bi1 = xBias[kOff + 2 * kp + 1];
      #pragma unroll
      for (int i = 0; i < 8; ++i)
        ((unsigned*)&tiles[buf].B[ng + i][0])[kp] =
            pack_bf16x2(pb0[i] * sc0 + bi0, pb1[i] * sc1 + bi1);
    } else {
      #pragma unroll
      for (int i = 0; i < 8; ++i)
        ((unsigned*)&tiles[buf].B[ng + i][0])[kp] = pack_bf16x2(pb0[i], pb1[i]);
    }
  };

  v8f acc[4] = {};
  const int arow = (wm << 4) + rl;
  const int nsteps = Kdim >> 5;

  { // prologue: stage k-step 0 into buffer 0
    float pa[8], pb0[8], pb1[8];
    #pragma unroll
    for (int i = 0; i < 8; ++i) pa[i] = baseA[i];
    const float* s0 = baseB;
    const float* s1 = baseB + N;
    #pragma unroll
    for (int i = 0; i < 8; ++i) { pb0[i] = s0[i]; pb1[i] = s1[i]; }
    store_tile(0, 0, pa, pb0, pb1);
  }
  __syncthreads();

  for (int s = 0; s < nsteps; ++s) {
    const int p = s & 1;
    // ---- prefetch next tile into registers (loads issue now, waited later) -
    float pa[8], pb0[8], pb1[8];
    const bool hasNext = (s + 1 < nsteps);
    if (hasNext) {
      const int k1 = (s + 1) << 5;
      const float* srcA = baseA + k1;
      #pragma unroll
      for (int i = 0; i < 8; ++i) pa[i] = srcA[i];
      const float* s0 = baseB + (size_t)k1 * N;
      const float* s1 = s0 + N;
      #pragma unroll
      for (int i = 0; i < 8; ++i) { pb0[i] = s0[i]; pb1[i] = s1[i]; }
    }
    // ---- compute on buffer p: issue ALL fragment loads, then 4 WMMAs -------
    {
      v16bf a  = join8(*(const v8bf*)&tiles[p].A[arow][kh * 8],
                       *(const v8bf*)&tiles[p].A[arow][kh * 8 + 16]);
      const int c0 = (wn << 6) + rl;
      v16bf b0 = join8(*(const v8bf*)&tiles[p].B[c0     ][kh * 8],
                       *(const v8bf*)&tiles[p].B[c0     ][kh * 8 + 16]);
      v16bf b1 = join8(*(const v8bf*)&tiles[p].B[c0 + 16][kh * 8],
                       *(const v8bf*)&tiles[p].B[c0 + 16][kh * 8 + 16]);
      v16bf b2 = join8(*(const v8bf*)&tiles[p].B[c0 + 32][kh * 8],
                       *(const v8bf*)&tiles[p].B[c0 + 32][kh * 8 + 16]);
      v16bf b3 = join8(*(const v8bf*)&tiles[p].B[c0 + 48][kh * 8],
                       *(const v8bf*)&tiles[p].B[c0 + 48][kh * 8 + 16]);
      SCHED_FENCE();  // keep all ds_loads above the WMMA burst -> partial waits
      acc[0] = __builtin_amdgcn_wmma_f32_16x16x32_bf16(false, a, false, b0, (short)0, acc[0], false, false);
      acc[1] = __builtin_amdgcn_wmma_f32_16x16x32_bf16(false, a, false, b1, (short)0, acc[1], false, false);
      acc[2] = __builtin_amdgcn_wmma_f32_16x16x32_bf16(false, a, false, b2, (short)0, acc[2], false, false);
      acc[3] = __builtin_amdgcn_wmma_f32_16x16x32_bf16(false, a, false, b3, (short)0, acc[3], false, false);
    }
    // ---- write prefetched tile into the other buffer -----------------------
    if (hasNext) store_tile(p ^ 1, (s + 1) << 5, pa, pb0, pb1);
    __syncthreads();
  }

  float alpha = 1.0f;
  if (MODE >= 1) alpha = *alphaPtr;
  #pragma unroll
  for (int nt = 0; nt < 4; ++nt) {
    const int n = n0 + (wn << 6) + (nt << 4) + rl;
    #pragma unroll
    for (int r = 0; r < 8; ++r) {
      const int m = m0 + (wm << 4) + (kh << 3) + r;
      float v = acc[nt][r];
      if (MODE == 2) v = 0.5f * v * (1.0f + erff(v * 0.70710678118654752f)); // exact GELU
      if (MODE >= 1) v = res[(size_t)b * resBatchStride + (size_t)m * N + n] + alpha * v;
      Out[(size_t)b * outBatchStride + (size_t)m * N + n] = v;
    }
  }
}

// ---------------------------------------------------------------------------
// Per-channel batch-norm stats over (N,H,W) -> folded scale/bias.
// ---------------------------------------------------------------------------
__global__ void __launch_bounds__(256)
bn_stats_kernel(const float* __restrict__ X, const float* __restrict__ gamma,
                const float* __restrict__ beta, float* __restrict__ scale,
                float* __restrict__ bias, long batchStride)
{
  const int N = HW_N;
  const int c = blockIdx.x;
  float s = 0.f, s2 = 0.f;
  for (int b = 0; b < NB; ++b) {
    const float* p = X + (size_t)b * batchStride + (size_t)c * N;
    for (int i = threadIdx.x; i < N; i += 256) { float v = p[i]; s += v; s2 += v * v; }
  }
  __shared__ float rs[256], rq[256];
  rs[threadIdx.x] = s; rq[threadIdx.x] = s2;
  __syncthreads();
  for (int st = 128; st > 0; st >>= 1) {
    if (threadIdx.x < st) {
      rs[threadIdx.x] += rs[threadIdx.x + st];
      rq[threadIdx.x] += rq[threadIdx.x + st];
    }
    __syncthreads();
  }
  if (threadIdx.x == 0) {
    const float inv = 1.0f / (float)((long)NB * N);
    float mean = rs[0] * inv;
    float var  = rq[0] * inv - mean * mean;
    float sc   = gamma[c] * rsqrtf(var + 1e-5f);
    scale[c] = sc;
    bias[c]  = beta[c] - mean * sc;
  }
}

// ---------------------------------------------------------------------------
// Windowed attention, one 128-thread block per (window, head).
// Tokens padded 49 -> 64; BN + 1/sqrt(Ch) folded into bf16 staging.
// S = Q K^T and O = P V both via WMMA; masked f32 softmax over 49 cols.
// CH = head dim: 64 (dir 1, qi=128) or 32 (dir 2, ki=64).
// ---------------------------------------------------------------------------
template<int CH>
__global__ void __launch_bounds__(128)
win_attn_wmma(const float* __restrict__ Q, const float* __restrict__ K,
              const float* __restrict__ V,
              const float* __restrict__ sQ, const float* __restrict__ bQ,
              const float* __restrict__ sK, const float* __restrict__ bK,
              const float* __restrict__ sV, const float* __restrict__ bV,
              float* __restrict__ O)
{
  const int C = 2 * CH;
  __shared__ __align__(16) bf16_t Qs[64][CH];   // [token][d], q*scale folded
  __shared__ __align__(16) bf16_t Ks[64][CH];   // [token][d] == Bt for QK^T
  __shared__ __align__(16) bf16_t Vt[CH][64];   // [d][token] == Bt for PV
  __shared__ __align__(16) float  Ss[64][64];   // scores (f32)
  __shared__ __align__(16) bf16_t Ps[64][64];   // probs (bf16)

  const int head = blockIdx.x & 1;
  const int wlin = blockIdx.x >> 1;
  const int b    = wlin >> 8;          // 16x16 = 256 windows per image
  const int wrem = wlin & 255;
  const int wh   = wrem >> 4, ww = wrem & 15;
  const float qscl = rsqrtf((float)CH);

  for (int e = threadIdx.x; e < 64 * CH; e += 128) {
    const int d   = e >> 6;
    const int tok = e & 63;
    const int c   = head * CH + d;
    float qv = 0.f, kv = 0.f, vv = 0.f;
    if (tok < 49) {
      const int y = wh * 7 + tok / 7;
      const int x = ww * 7 + tok % 7;
      const size_t idx = ((size_t)(b * C + c)) * HW_N + y * IMG_W + x;
      qv = (Q[idx] * sQ[c] + bQ[c]) * qscl;
      kv =  K[idx] * sK[c] + bK[c];
      vv =  V[idx] * sV[c] + bV[c];
    }
    Qs[tok][d] = (bf16_t)qv;
    Ks[tok][d] = (bf16_t)kv;
    Vt[d][tok] = (bf16_t)vv;
  }
  __syncthreads();

  const int wave = threadIdx.x >> 5;   // wave w -> S/O rows [16w, 16w+16)
  const int lane = threadIdx.x & 31;
  const int rl   = lane & 15;
  const int kh   = lane >> 4;
  const int arow = (wave << 4) + rl;

  // ---- S = Q K^T -----------------------------------------------------------
  v8f sacc[4] = {};
  #pragma unroll
  for (int k0 = 0; k0 < CH; k0 += 32) {
    v16bf a = join8(*(const v8bf*)&Qs[arow][k0 + kh * 8],
                    *(const v8bf*)&Qs[arow][k0 + kh * 8 + 16]);
    v16bf bb[4];
    #pragma unroll
    for (int nt = 0; nt < 4; ++nt) {
      const int col = (nt << 4) + rl;
      bb[nt] = join8(*(const v8bf*)&Ks[col][k0 + kh * 8],
                     *(const v8bf*)&Ks[col][k0 + kh * 8 + 16]);
    }
    SCHED_FENCE();
    #pragma unroll
    for (int nt = 0; nt < 4; ++nt)
      sacc[nt] = __builtin_amdgcn_wmma_f32_16x16x32_bf16(
          false, a, false, bb[nt], (short)0, sacc[nt], false, false);
  }
  #pragma unroll
  for (int nt = 0; nt < 4; ++nt)
    #pragma unroll
    for (int r = 0; r < 8; ++r)
      Ss[(wave << 4) + (kh << 3) + r][(nt << 4) + rl] = sacc[nt][r];
  __syncthreads();

  // ---- masked softmax over the 49 real columns -----------------------------
  if (threadIdx.x < 64) {
    const int r = threadIdx.x;
    float mx = -1e30f;
    for (int c2 = 0; c2 < 49; ++c2) mx = fmaxf(mx, Ss[r][c2]);
    float sum = 0.f;
    for (int c2 = 0; c2 < 49; ++c2) { float ev = __expf(Ss[r][c2] - mx); Ss[r][c2] = ev; sum += ev; }
    const float inv = 1.0f / sum;
    for (int c2 = 0; c2 < 49; ++c2) Ps[r][c2] = (bf16_t)(Ss[r][c2] * inv);
    for (int c2 = 49; c2 < 64; ++c2) Ps[r][c2] = (bf16_t)0.0f;
  }
  __syncthreads();

  // ---- O = P V -------------------------------------------------------------
  constexpr int NT = CH / 16;
  v8f oacc[NT] = {};
  #pragma unroll
  for (int k0 = 0; k0 < 64; k0 += 32) {
    v16bf a = join8(*(const v8bf*)&Ps[arow][k0 + kh * 8],
                    *(const v8bf*)&Ps[arow][k0 + kh * 8 + 16]);
    v16bf bb[NT];
    #pragma unroll
    for (int nt = 0; nt < NT; ++nt) {
      const int col = (nt << 4) + rl;
      bb[nt] = join8(*(const v8bf*)&Vt[col][k0 + kh * 8],
                     *(const v8bf*)&Vt[col][k0 + kh * 8 + 16]);
    }
    SCHED_FENCE();
    #pragma unroll
    for (int nt = 0; nt < NT; ++nt)
      oacc[nt] = __builtin_amdgcn_wmma_f32_16x16x32_bf16(
          false, a, false, bb[nt], (short)0, oacc[nt], false, false);
  }
  #pragma unroll
  for (int nt = 0; nt < NT; ++nt) {
    #pragma unroll
    for (int r = 0; r < 8; ++r) {
      const int l = (wave << 4) + (kh << 3) + r;
      if (l < 49) {
        const int d = (nt << 4) + rl;
        const int c = head * CH + d;
        const int y = wh * 7 + l / 7;
        const int x = ww * 7 + l % 7;
        O[((size_t)(b * C + c)) * HW_N + y * IMG_W + x] = oacc[nt][r];
      }
    }
  }
}

// ---------------------------------------------------------------------------
extern "C" void kernel_launch(void* const* d_in, const int* in_sizes, int n_in,
                              void* d_out, int out_size, void* d_ws, size_t ws_size,
                              hipStream_t stream) {
  (void)in_sizes; (void)n_in; (void)out_size; (void)ws_size;
  const float* q_src   = (const float*)d_in[0];
  const float* kv_src  = (const float*)d_in[1];
  const float* w_q1    = (const float*)d_in[2];
  const float* w_k1    = (const float*)d_in[3];
  const float* w_v1    = (const float*)d_in[4];
  const float* w_o1    = (const float*)d_in[5];
  const float* w_q2    = (const float*)d_in[6];
  const float* w_k2    = (const float*)d_in[7];
  const float* w_v2    = (const float*)d_in[8];
  const float* w_o2    = (const float*)d_in[9];
  const float* w_kv2q  = (const float*)d_in[10];
  const float* w_fuse  = (const float*)d_in[11];
  const float* g_q1 = (const float*)d_in[12]; const float* b_q1 = (const float*)d_in[13];
  const float* g_k1 = (const float*)d_in[14]; const float* b_k1 = (const float*)d_in[15];
  const float* g_v1 = (const float*)d_in[16]; const float* b_v1 = (const float*)d_in[17];
  const float* g_q2 = (const float*)d_in[18]; const float* b_q2 = (const float*)d_in[19];
  const float* g_k2 = (const float*)d_in[20]; const float* b_k2 = (const float*)d_in[21];
  const float* g_v2 = (const float*)d_in[22]; const float* b_v2 = (const float*)d_in[23];
  const float* g_fu = (const float*)d_in[24]; const float* b_fu = (const float*)d_in[25];
  const float* gamma_q  = (const float*)d_in[26];
  const float* gamma_kv = (const float*)d_in[27];
  const float* delta    = (const float*)d_in[28];
  float* out = (float*)d_out;

  const long N = HW_N;
  float* ws = (float*)d_ws;
  size_t off = 0;
  auto carve = [&](size_t n) { float* p = ws + off; off += n; return p; };

  float* q1    = carve((size_t)NB * 128 * N);
  float* k1    = carve((size_t)NB * 128 * N);
  float* v1    = carve((size_t)NB * 128 * N);
  float* q2    = carve((size_t)NB * 64  * N);
  float* k2    = carve((size_t)NB * 64  * N);
  float* v2    = carve((size_t)NB * 64  * N);
  float* aO1   = carve((size_t)NB * 128 * N);   // win_attn output dir 1
  float* aO2   = carve((size_t)NB * 64  * N);   // win_attn output dir 2
  float* kvEnh = carve((size_t)NB * 128 * N);   // kv_src + gamma_kv * attn_kv
  float* cat   = carve((size_t)NB * 512 * N);   // [q_enh ; kv_in_q]
  float* sc_q1 = carve(128); float* bi_q1 = carve(128);
  float* sc_k1 = carve(128); float* bi_k1 = carve(128);
  float* sc_v1 = carve(128); float* bi_v1 = carve(128);
  float* sc_q2 = carve(64);  float* bi_q2 = carve(64);
  float* sc_k2 = carve(64);  float* bi_k2 = carve(64);
  float* sc_v2 = carve(64);  float* bi_v2 = carve(64);
  float* sc_fu = carve(512); float* bi_fu = carve(512);

  const dim3 blk(256);
  // ---- projection GEMMs (raw conv outputs; BN applied downstream) ----------
  gemm_conv1_wmma<0, false><<<dim3(98, 2, NB), blk, 0, stream>>>(w_q1, q_src,  q1, nullptr, nullptr, nullptr, nullptr, 256, 256 * N, 128 * N, 0);
  gemm_conv1_wmma<0, false><<<dim3(98, 2, NB), blk, 0, stream>>>(w_k1, kv_src, k1, nullptr, nullptr, nullptr, nullptr, 128, 128 * N, 128 * N, 0);
  gemm_conv1_wmma<0, false><<<dim3(98, 2, NB), blk, 0, stream>>>(w_v1, kv_src, v1, nullptr, nullptr, nullptr, nullptr, 128, 128 * N, 128 * N, 0);
  gemm_conv1_wmma<0, false><<<dim3(98, 1, NB), blk, 0, stream>>>(w_q2, kv_src, q2, nullptr, nullptr, nullptr, nullptr, 128, 128 * N, 64 * N, 0);
  gemm_conv1_wmma<0, false><<<dim3(98, 1, NB), blk, 0, stream>>>(w_k2, q_src,  k2, nullptr, nullptr, nullptr, nullptr, 256, 256 * N, 64 * N, 0);
  gemm_conv1_wmma<0, false><<<dim3(98, 1, NB), blk, 0, stream>>>(w_v2, q_src,  v2, nullptr, nullptr, nullptr, nullptr, 256, 256 * N, 64 * N, 0);

  // ---- BN stats (folded scale/bias) ----------------------------------------
  bn_stats_kernel<<<128, blk, 0, stream>>>(q1, g_q1, b_q1, sc_q1, bi_q1, 128 * N);
  bn_stats_kernel<<<128, blk, 0, stream>>>(k1, g_k1, b_k1, sc_k1, bi_k1, 128 * N);
  bn_stats_kernel<<<128, blk, 0, stream>>>(v1, g_v1, b_v1, sc_v1, bi_v1, 128 * N);
  bn_stats_kernel<<<64,  blk, 0, stream>>>(q2, g_q2, b_q2, sc_q2, bi_q2, 64 * N);
  bn_stats_kernel<<<64,  blk, 0, stream>>>(k2, g_k2, b_k2, sc_k2, bi_k2, 64 * N);
  bn_stats_kernel<<<64,  blk, 0, stream>>>(v2, g_v2, b_v2, sc_v2, bi_v2, 64 * N);

  // ---- windowed attention (8 imgs * 256 windows * 2 heads = 4096 blocks) ---
  win_attn_wmma<64><<<4096, 128, 0, stream>>>(q1, k1, v1, sc_q1, bi_q1, sc_k1, bi_k1, sc_v1, bi_v1, aO1);
  win_attn_wmma<32><<<4096, 128, 0, stream>>>(q2, k2, v2, sc_q2, bi_q2, sc_k2, bi_k2, sc_v2, bi_v2, aO2);

  // ---- output projections with fused residual + gamma ----------------------
  // q_enh -> rows [0,256) of cat ; kv_enh -> own buffer
  gemm_conv1_wmma<1, false><<<dim3(98, 4, NB), blk, 0, stream>>>(w_o1, aO1, cat,   nullptr, nullptr, q_src,  gamma_q,  128, 128 * N, 512 * N, 256 * N);
  gemm_conv1_wmma<1, false><<<dim3(98, 2, NB), blk, 0, stream>>>(w_o2, aO2, kvEnh, nullptr, nullptr, kv_src, gamma_kv, 64,  64  * N, 128 * N, 128 * N);

  // ---- kv_in_q = conv1(kv_enh, w_kv2q) -> rows [256,512) of cat ------------
  gemm_conv1_wmma<0, false><<<dim3(98, 4, NB), blk, 0, stream>>>(w_kv2q, kvEnh, cat + 256 * N, nullptr, nullptr, nullptr, nullptr, 128, 128 * N, 512 * N, 0);

  // ---- fuse: BN(cat) folded into staging, GELU+residual+delta in epilogue --
  bn_stats_kernel<<<512, blk, 0, stream>>>(cat, g_fu, b_fu, sc_fu, bi_fu, 512 * N);
  gemm_conv1_wmma<2, true><<<dim3(98, 4, NB), blk, 0, stream>>>(w_fuse, cat, out, sc_fu, bi_fu, q_src, delta, 512, 512 * N, 256 * N, 256 * N);
}
